// GATEncoder_7842610283172
// MI455X (gfx1250) — matrix operand order repaired and checked by reading, not compiled, not run on hardware
//
#include <hip/hip_runtime.h>
#include <hip/hip_bf16.h>

// TransformerConv (GAT-style) x3 layers for gfx1250 (MI455X).
// N=10000, C=256, H=8, D=32, E=320000, L=3.
// Dense projections: bf16 WMMA (v_wmma_f32_16x16x32_bf16). One 512-thread
// block per 16-row tile: the A tile (16x256 bf16, 8KB) is staged into LDS
// once via global_load_async_to_lds_b128 (ASYNCcnt), then 16 waves
// (4 matrices x 4 column groups) each compute a 16x64 tile with
// ds_load_b128 A-fragments and global_load_b128 B-fragments.
// Edge softmax/aggregation: L2-resident gathers + f32 atomics.

#define CC   256
#define HH   8
#define DD   32
#define LL   3
#define INV_SQRT_D 0.17677669529663687f   // 1/sqrt(32)
#define NEG_INF_ENC 0x007FFFFFu           // monotone-encoded -inf

typedef __attribute__((ext_vector_type(16))) __bf16  v16bf;
typedef __attribute__((ext_vector_type(8)))  float   v8f;
typedef __attribute__((ext_vector_type(8)))  unsigned v8u;

// float -> bf16 bits, round-to-nearest-even
__device__ __forceinline__ unsigned short f2bf(float f) {
    unsigned u = __float_as_uint(f);
    unsigned r = u + 0x7FFFu + ((u >> 16) & 1u);
    return (unsigned short)(r >> 16);
}

// order-preserving float <-> uint encoding (for atomicMax on floats)
__device__ __forceinline__ unsigned fenc(float f) {
    unsigned u = __float_as_uint(f);
    return (u & 0x80000000u) ? ~u : (u | 0x80000000u);
}
__device__ __forceinline__ float fdec(unsigned u) {
    unsigned v = (u & 0x80000000u) ? (u & 0x7FFFFFFFu) : ~u;
    return __uint_as_float(v);
}

__device__ __forceinline__ v16bf frag_from(const unsigned short* p0,
                                           const unsigned short* p1) {
    const uint4 a = *(const uint4*)p0;
    const uint4 b = *(const uint4*)p1;
    v8u u = {a.x, a.y, a.z, a.w, b.x, b.y, b.z, b.w};
    return __builtin_bit_cast(v16bf, u);
}

// ---------------------------------------------------------------------------
// per-layer init: m_enc = enc(-inf), denom = 0      (one thread per (n,h))
// ---------------------------------------------------------------------------
__global__ void tc_init_kernel(unsigned* __restrict__ menc,
                               float* __restrict__ denom, int nh) {
    int i = blockIdx.x * blockDim.x + threadIdx.x;
    if (i < nh) { menc[i] = NEG_INF_ENC; denom[i] = 0.0f; }
}

// ---------------------------------------------------------------------------
// x (f32, row-major) -> bf16 row-major
// ---------------------------------------------------------------------------
__global__ void tc_cvt_x(const float* __restrict__ x,
                         unsigned short* __restrict__ xb, int n) {
    int i = blockIdx.x * blockDim.x + threadIdx.x;
    if (i < n) xb[i] = f2bf(x[i]);
}

// ---------------------------------------------------------------------------
// {Wq,Wk,Wv,Ws} (f32, [k][n]) -> bf16 transposed Wt[mat][n][k]
// ---------------------------------------------------------------------------
__global__ void tc_cvt_w(const float* __restrict__ Wq, const float* __restrict__ Wk,
                         const float* __restrict__ Wv, const float* __restrict__ Ws,
                         unsigned short* __restrict__ wt) {
    int i = blockIdx.x * blockDim.x + threadIdx.x;     // < 4*CC*CC
    const int mat = i >> 16;                            // CC*CC == 65536
    const int rem = i & 0xFFFF;
    const int kr  = rem >> 8;                           // row of W (input dim)
    const int n   = rem & 255;                          // col of W (output dim)
    const float* W = (mat == 0) ? Wq : (mat == 1) ? Wk : (mat == 2) ? Wv : Ws;
    wt[(size_t)mat * CC * CC + (size_t)n * CC + kr] = f2bf(W[(size_t)kr * CC + n]);
}

// ---------------------------------------------------------------------------
// fused projection GEMM: out = x @ W + b for all W in {Wq,Wk,Wv,Ws}.
// grid = (N/16), block = (32,16): wave w -> matrix w>>2, column group w&3.
// A tile staged once per block into LDS via global_load_async_to_lds_b128.
// ---------------------------------------------------------------------------
__global__ void __launch_bounds__(512)
tc_proj_gemm(const unsigned short* __restrict__ xb,
             const unsigned short* __restrict__ wt,
             const float* __restrict__ bq, const float* __restrict__ bk,
             const float* __restrict__ bv, const float* __restrict__ bs,
             float* __restrict__ qo, float* __restrict__ ko,
             float* __restrict__ vo, float* __restrict__ so,
             int Nn) {
    __shared__ unsigned short As[16 * CC];              // 8 KB A tile

    const int lane = threadIdx.x;          // 0..31
    const int wave = threadIdx.y;          // 0..15
    const int tid  = wave * 32 + lane;     // 0..511
    const int half = lane >> 4;            // 0/1
    const int m    = lane & 15;
    const int row0 = blockIdx.x * 16;
    const bool full = (row0 + 16 <= Nn);   // always true for N % 16 == 0

    // ---- stage A tile (16 x 256 bf16 = 8192 B): one b128 per lane ----
    if (full) {
        const unsigned byte_off = (unsigned)tid * 16u;
        unsigned lds_addr = (unsigned)(uintptr_t)(&As[0]) + byte_off;
        unsigned long long gaddr =
            (unsigned long long)(uintptr_t)(xb + (size_t)row0 * CC) + byte_off;
        asm volatile("global_load_async_to_lds_b128 %0, %1, off"
                     :: "v"(lds_addr), "v"(gaddr) : "memory");
        asm volatile("s_wait_asynccnt 0x0" ::: "memory");
    } else {
        for (int i = tid; i < 16 * CC; i += 512) {
            int row = row0 + (i >> 8); if (row >= Nn) row = Nn - 1;
            As[i] = xb[(size_t)row * CC + (i & 255)];
        }
    }
    __syncthreads();

    const int mat  = wave >> 2;            // 0..3 -> q,k,v,skip
    const int col0 = (wave & 3) * 64;      // 0,64,128,192

    const float* bias; float* dstp;
    switch (mat) {
        case 0:  bias = bq; dstp = qo; break;
        case 1:  bias = bk; dstp = ko; break;
        case 2:  bias = bv; dstp = vo; break;
        default: bias = bs; dstp = so; break;
    }
    const unsigned short* wtm = wt + (size_t)mat * CC * CC;

    // A fragment source in LDS: lane m's row, K window base 8*half
    const unsigned short* abase = &As[m * CC + 8 * half];

    // B columns: Wt[col][k], K window base 8*half
    const unsigned short* wbase[4];
    #pragma unroll
    for (int t = 0; t < 4; ++t)
        wbase[t] = wtm + (size_t)(col0 + t * 16 + m) * CC + 8 * half;

    v8f acc[4];
    #pragma unroll
    for (int t = 0; t < 4; ++t) acc[t] = (v8f){0.f,0.f,0.f,0.f,0.f,0.f,0.f,0.f};

    for (int kk = 0; kk < CC; kk += 32) {
        // A frag from LDS (ds_load_b128 x2)
        const v16bf A = frag_from(abase + kk, abase + kk + 16);
        #pragma unroll
        for (int t = 0; t < 4; ++t) {
            // B frag: 32 contiguous bf16 along K in the transposed weight row
            const v16bf B = frag_from(wbase[t] + kk, wbase[t] + kk + 16);
            acc[t] = __builtin_amdgcn_wmma_f32_16x16x32_bf16(
                false, A, false, B, (short)0, acc[t], false, false);
        }
    }

    // D store: VGPR r -> (M = 8*half + r, N = lane&15); uniform tail branch
    if (full) {
        #pragma unroll
        for (int t = 0; t < 4; ++t) {
            const int col = col0 + t * 16 + m;
            const float bv_ = bias[col];
            float* dcol = dstp + (size_t)(row0 + 8 * half) * CC + col;
            #pragma unroll
            for (int r = 0; r < 8; ++r)
                dcol[(size_t)r * CC] = acc[t][r] + bv_;
        }
    } else {
        #pragma unroll
        for (int t = 0; t < 4; ++t) {
            const int col = col0 + t * 16 + m;
            const float bv_ = bias[col];
            #pragma unroll
            for (int r = 0; r < 8; ++r) {
                const int row = row0 + 8 * half + r;
                if (row < Nn) dstp[(size_t)row * CC + col] = acc[t][r] + bv_;
            }
        }
    }
}

// ---------------------------------------------------------------------------
// per-edge attention scores + scatter-max        (one thread per (e,h))
// ---------------------------------------------------------------------------
__global__ void tc_score_kernel(const float* __restrict__ q, const float* __restrict__ k,
                                const int* __restrict__ src, const int* __restrict__ dst,
                                float* __restrict__ score, unsigned* __restrict__ menc,
                                int eh) {
    int idx = blockIdx.x * blockDim.x + threadIdx.x;
    if (idx >= eh) return;
    const int e = idx >> 3, h = idx & 7;
    const int s = src[e], d = dst[e];
    const float4* qp = (const float4*)(q + (size_t)d * CC + h * DD);
    const float4* kp = (const float4*)(k + (size_t)s * CC + h * DD);
    float acc = 0.0f;
    #pragma unroll
    for (int i = 0; i < DD / 4; ++i) {
        const float4 a = qp[i], b = kp[i];
        acc += a.x * b.x + a.y * b.y + a.z * b.z + a.w * b.w;
    }
    acc *= INV_SQRT_D;
    score[idx] = acc;
    atomicMax(&menc[d * HH + h], fenc(acc));
}

// ---------------------------------------------------------------------------
// a = exp(score - m[dst]); denom[dst] += a        (one thread per (e,h))
// ---------------------------------------------------------------------------
__global__ void tc_exp_kernel(float* __restrict__ score,
                              const unsigned* __restrict__ menc,
                              float* __restrict__ denom,
                              const int* __restrict__ dst, int eh) {
    int idx = blockIdx.x * blockDim.x + threadIdx.x;
    if (idx >= eh) return;
    const int e = idx >> 3, h = idx & 7;
    const int d = dst[e];
    const float a = __expf(score[idx] - fdec(menc[d * HH + h]));
    score[idx] = a;
    atomicAdd(&denom[d * HH + h], a);
}

// ---------------------------------------------------------------------------
// out[dst] += (a/denom[dst]) * v[src]   (block per edge, thread per channel)
// out already holds the skip term x@Ws+bs from tc_proj_gemm.
// ---------------------------------------------------------------------------
__global__ void tc_agg_kernel(const float* __restrict__ score,
                              const float* __restrict__ denom,
                              const float* __restrict__ v,
                              const int* __restrict__ src, const int* __restrict__ dst,
                              float* __restrict__ out, int Ee) {
    const int e = blockIdx.x;
    if (e >= Ee) return;
    const int c = threadIdx.x;             // 0..255
    const int h = c >> 5;
    const int s = src[e], d = dst[e];
    const float a  = score[e * HH + h];
    const float dn = denom[d * HH + h];
    const float alpha = a / (dn + 1e-16f);
    atomicAdd(&out[(size_t)d * CC + c], alpha * v[(size_t)s * CC + c]);
}

// ---------------------------------------------------------------------------
extern "C" void kernel_launch(void* const* d_in, const int* in_sizes, int n_in,
                              void* d_out, int out_size, void* d_ws, size_t ws_size,
                              hipStream_t stream) {
    const float* x  = (const float*)d_in[0];
    const float* Wq = (const float*)d_in[1];
    const float* bq = (const float*)d_in[2];
    const float* Wk = (const float*)d_in[3];
    const float* bk = (const float*)d_in[4];
    const float* Wv = (const float*)d_in[5];
    const float* bv = (const float*)d_in[6];
    const float* Ws = (const float*)d_in[7];
    const float* bs = (const float*)d_in[8];
    const int*   ei = (const int*)d_in[9];

    const int Nn = in_sizes[0] / CC;       // 10000
    const int Ee = in_sizes[9] / 2;        // 320000
    const int* src = ei;
    const int* dst = ei + Ee;

    // scratch carve-up; ~68 MB total
    const size_t nc = (size_t)Nn * CC;
    float* p = (float*)d_ws;
    float*    qb    = p;             p += nc;
    float*    kb    = p;             p += nc;
    float*    vb    = p;             p += nc;
    float*    sc    = p;             p += (size_t)Ee * HH;
    unsigned* menc  = (unsigned*)p;  p += (size_t)Nn * HH;
    float*    denom = p;             p += (size_t)Nn * HH;
    float*    buf0  = p;             p += nc;
    float*    buf1  = p;             p += nc;
    unsigned short* xbb = (unsigned short*)p;  p += nc / 2;            // bf16 x
    unsigned short* wtb = (unsigned short*)p;  p += (4 * CC * CC) / 2; // bf16 W^T

    float* outs[LL] = { buf0, buf1, (float*)d_out };

    const int nh = Nn * HH;
    const int eh = Ee * HH;
    const int ncl = (int)nc;
    const dim3 gemm_grid((unsigned)((Nn + 15) / 16), 1, 1);
    const dim3 gemm_block(32, 16);

    for (int l = 0; l < LL; ++l) {
        const float* xl = (l == 0) ? x : outs[l - 1];
        float* ol = outs[l];
        const size_t wOff = (size_t)l * CC * CC;
        const size_t bOff = (size_t)l * CC;

        tc_init_kernel<<<(nh + 255) / 256, 256, 0, stream>>>(menc, denom, nh);

        tc_cvt_x<<<(ncl + 255) / 256, 256, 0, stream>>>(xl, xbb, ncl);
        tc_cvt_w<<<(4 * CC * CC) / 256, 256, 0, stream>>>(
            Wq + wOff, Wk + wOff, Wv + wOff, Ws + wOff, wtb);

        tc_proj_gemm<<<gemm_grid, gemm_block, 0, stream>>>(
            xbb, wtb, bq + bOff, bk + bOff, bv + bOff, bs + bOff,
            qb, kb, vb, ol, Nn);

        tc_score_kernel<<<(eh + 255) / 256, 256, 0, stream>>>(
            qb, kb, src, dst, sc, menc, eh);

        tc_exp_kernel<<<(eh + 255) / 256, 256, 0, stream>>>(
            sc, menc, denom, dst, eh);

        tc_agg_kernel<<<Ee, CC, 0, stream>>>(
            sc, denom, vb, src, dst, ol, Ee);
    }
}
